// QuantumInference_55473797595916
// MI455X (gfx1250) — matrix operand order repaired and checked by reading, compiled, not run on hardware
//
#include <hip/hip_runtime.h>

#define E 512
#define Q 64
#define NROWS (32*4096)
#define TOTAL ((size_t)NROWS * Q)
#define PI_F 3.14159265358979323846f

typedef __attribute__((ext_vector_type(16))) __bf16 v16bf;
typedef __attribute__((ext_vector_type(8)))  float  v8f;
typedef __attribute__((ext_vector_type(4)))  float  f4;

// d_ws layout (ushort == bf16 bits). Big mats: 64 blocks of 512 elems each
// (block = kstep*4 + ntile, elem = lane*16 + e). Small mats: 8 blocks.
#define WR_OFF   0
#define WI_OFF   32768
#define WP1_OFF  65536
#define WP2_OFF  98304
#define ROP_OFF  102400
#define IOP_OFF  106496
#define WS_ELEMS 110592

__device__ __forceinline__ unsigned short bfbits(float x) {
  union { __bf16 h; unsigned short s; } u;
  u.h = (__bf16)x;            // RNE hardware convert
  return u.s;
}

union BFV {
  unsigned short s[16];
  uint4 u4[2];
  v16bf v;
};

// ---------------- prep: weights + Hermitian operator -> bf16 fragment layout ----------------
__global__ void qi_prep(const float* __restrict__ Wr, const float* __restrict__ Wi,
                        const float* __restrict__ Wp1, const float* __restrict__ Wp2,
                        const float* __restrict__ op, unsigned short* __restrict__ ws)
{
  int idx = blockIdx.x * 256 + threadIdx.x;
  if (idx >= WS_ELEMS) return;
  float val;
  if (idx < WP2_OFF) {
    int mat = idx / 32768, r = idx % 32768;
    int block = r / 512, off = r % 512;
    int lane = off / 16, e = off % 16;
    int kstep = block / 4, n = block % 4;
    int K   = kstep * 32 + (lane >> 4) * 8 + (e & 7) + (e >> 3) * 16;   // B frag: lanes<16 K={0..7,16..23}, lanes>=16 K={8..15,24..31}
    int col = n * 16 + (lane & 15);
    const float* W = (mat == 0) ? Wr : (mat == 1) ? Wi : Wp1;
    val = W[K * Q + col];
  } else {
    int r2 = idx - WP2_OFF;
    int mat = r2 / 4096, rr = r2 % 4096;
    int block = rr / 512, off = rr % 512;
    int lane = off / 16, e = off % 16;
    int kstep = block / 4, n = block % 4;
    int K   = kstep * 32 + (lane >> 4) * 8 + (e & 7) + (e >> 3) * 16;
    int col = n * 16 + (lane & 15);
    if (mat == 0)      val = Wp2[K * Q + col];
    else if (mat == 1) val = (K == col) ? (1.0f / (1.0f + __expf(-op[K]))) : op[K] * op[col];
    else               val = op[K] - op[col];
  }
  ws[idx] = bfbits(val);
}

// ---------------- main ----------------
__device__ __forceinline__ v16bf loadB(const unsigned short* __restrict__ base, int lane) {
  BFV b;
  const uint4* p = (const uint4*)base + lane * 2;
  b.u4[0] = p[0];
  b.u4[1] = p[1];
  return b.v;
}

__device__ __forceinline__ v8f zero8() {
  v8f z;
  #pragma unroll
  for (int j = 0; j < 8; ++j) z[j] = 0.0f;
  return z;
}

#define WMMA_BF16(A, B, C) \
  __builtin_amdgcn_wmma_f32_16x16x32_bf16(false, (A), false, (B), (short)0, (C), false, false)

__global__ __launch_bounds__(256) void qi_main(
    const float* __restrict__ feat,
    const float* __restrict__ b_real, const float* __restrict__ b_imag,
    const float* __restrict__ b_p1,   const float* __restrict__ b_p2,
    const unsigned short* __restrict__ ws, float* __restrict__ out)
{
  // per-wave staging tiles: row-major 16 x 64 bf16, row stride 72 (conflict-free ds_b128)
  __shared__ __align__(16) unsigned short tile[8][2][16 * 72];

  const int tid  = threadIdx.x;
  const int wave = tid >> 5;
  const int lane = tid & 31;
  const int l15  = lane & 15;
  const int hi   = lane >> 4;
  const int rowBase = blockIdx.x * 128 + wave * 16;

  // per-lane biases for the 4 N-tiles (C layout: col = n*16 + l15)
  float brv[4], biv[4], bpv[4], b2v[4];
  #pragma unroll
  for (int n = 0; n < 4; ++n) {
    int c = n * 16 + l15;
    brv[n] = b_real[c]; biv[n] = b_imag[c]; bpv[n] = b_p1[c]; b2v[n] = b_p2[c];
  }

  v8f ar[4], ai[4], ap[4];
  #pragma unroll
  for (int n = 0; n < 4; ++n) { ar[n] = zero8(); ai[n] = zero8(); ap[n] = zero8(); }

  // --------- big GEMMs: [16x512] x [512x64] for W_real/W_imag/W_p1 (shared A) ---------
  // Features are 268MB read-once: stream with non-temporal hints so the
  // L2-resident ws weights (re-read by all 1024 blocks) are not evicted.
  const float* frow = feat + (size_t)(rowBase + l15) * E;
  for (int ks = 0; ks < 16; ++ks) {
    const int k0 = ks * 32;
    // A frag: lanes<16 K = k0+{0..7,16..23}, lanes>=16 K = k0+{8..15,24..31}
    const f4* p0 = (const f4*)(frow + k0 + hi * 8);
    const f4* p1 = (const f4*)(frow + k0 + 16 + hi * 8);
    f4 c0 = __builtin_nontemporal_load(p0);
    f4 c1 = __builtin_nontemporal_load(p0 + 1);
    f4 c2 = __builtin_nontemporal_load(p1);
    f4 c3 = __builtin_nontemporal_load(p1 + 1);
    if (ks < 15) __builtin_prefetch(frow + k0 + 32 + hi * 8, 0, 3);
    v16bf av;
    av[ 0] = (__bf16)c0[0]; av[ 1] = (__bf16)c0[1]; av[ 2] = (__bf16)c0[2]; av[ 3] = (__bf16)c0[3];
    av[ 4] = (__bf16)c1[0]; av[ 5] = (__bf16)c1[1]; av[ 6] = (__bf16)c1[2]; av[ 7] = (__bf16)c1[3];
    av[ 8] = (__bf16)c2[0]; av[ 9] = (__bf16)c2[1]; av[10] = (__bf16)c2[2]; av[11] = (__bf16)c2[3];
    av[12] = (__bf16)c3[0]; av[13] = (__bf16)c3[1]; av[14] = (__bf16)c3[2]; av[15] = (__bf16)c3[3];
    #pragma unroll
    for (int n = 0; n < 4; ++n) {
      const int blk = (ks * 4 + n) * 512;
      v16bf bR = loadB(ws + WR_OFF  + blk, lane);
      v16bf bI = loadB(ws + WI_OFF  + blk, lane);
      v16bf bP = loadB(ws + WP1_OFF + blk, lane);
      ar[n] = WMMA_BF16(av, bR, ar[n]);
      ai[n] = WMMA_BF16(av, bI, ai[n]);
      ap[n] = WMMA_BF16(av, bP, ap[n]);
    }
  }

  // --------- phases = tanh(tanh(p1) @ W_p2 + b_p2) * pi ---------
  unsigned short* t0 = &tile[wave][0][0];
  unsigned short* t1 = &tile[wave][1][0];
  #pragma unroll
  for (int n = 0; n < 4; ++n)
    #pragma unroll
    for (int i = 0; i < 8; ++i) {
      float p = tanhf(ap[n][i] + bpv[n]);
      t0[(i + 8 * hi) * 72 + n * 16 + l15] = bfbits(p);   // C layout -> row-major tile
    }
  // per-wave DS ops are in-order; reload in A layout
  v16bf pa[2];
  #pragma unroll
  for (int kk = 0; kk < 2; ++kk) {
    BFV a;
    a.u4[0] = *(const uint4*)&t0[l15 * 72 + kk * 32 + hi * 8];
    a.u4[1] = *(const uint4*)&t0[l15 * 72 + kk * 32 + 16 + hi * 8];
    pa[kk] = a.v;
  }
  v8f aph[4];
  #pragma unroll
  for (int n = 0; n < 4; ++n) aph[n] = zero8();
  #pragma unroll
  for (int kk = 0; kk < 2; ++kk)
    #pragma unroll
    for (int n = 0; n < 4; ++n) {
      v16bf b = loadB(ws + WP2_OFF + (kk * 4 + n) * 512, lane);
      aph[n] = WMMA_BF16(pa[kk], b, aph[n]);
    }

  // --------- rotate by phase, L2-normalize ---------
  float rp[4][8], ipv[4][8], ssq[8];
  #pragma unroll
  for (int i = 0; i < 8; ++i) ssq[i] = 0.0f;
  #pragma unroll
  for (int n = 0; n < 4; ++n)
    #pragma unroll
    for (int i = 0; i < 8; ++i) {
      float ph = tanhf(aph[n][i] + b2v[n]) * PI_F;
      size_t ro = (size_t)(rowBase + i + 8 * hi) * Q + n * 16 + l15;
      __builtin_nontemporal_store(ph, &out[2 * TOTAL + ro]);      // phases output
      float cr = __cosf(ph), ci = __sinf(ph);
      float rr = tanhf(ar[n][i] + brv[n]);
      float mm = tanhf(ai[n][i] + biv[n]);
      float x = rr * cr - mm * ci;
      float y = rr * ci + mm * cr;
      rp[n][i] = x; ipv[n][i] = y;
      ssq[i] += x * x + y * y;
    }
  float inv[8];
  #pragma unroll
  for (int i = 0; i < 8; ++i) {
    float s = ssq[i];
    s += __shfl_xor(s, 1, 32);
    s += __shfl_xor(s, 2, 32);
    s += __shfl_xor(s, 4, 32);
    s += __shfl_xor(s, 8, 32);   // masks < 16: reduction stays in each 16-lane half (per-row)
    inv[i] = rsqrtf(s + 1e-12f);
  }

  // --------- operator apply: out_r = rn@R - inn@I, out_i = rn@I + inn@R ---------
  #pragma unroll
  for (int n = 0; n < 4; ++n)
    #pragma unroll
    for (int i = 0; i < 8; ++i) {
      int o = (i + 8 * hi) * 72 + n * 16 + l15;
      t0[o] = bfbits(rp[n][i]  * inv[i]);
      t1[o] = bfbits(ipv[n][i] * inv[i]);
    }
  v16bf ra[2], ina[2];
  #pragma unroll
  for (int kk = 0; kk < 2; ++kk) {
    BFV a, b;
    a.u4[0] = *(const uint4*)&t0[l15 * 72 + kk * 32 + hi * 8];
    a.u4[1] = *(const uint4*)&t0[l15 * 72 + kk * 32 + 16 + hi * 8];
    b.u4[0] = *(const uint4*)&t1[l15 * 72 + kk * 32 + hi * 8];
    b.u4[1] = *(const uint4*)&t1[l15 * 72 + kk * 32 + 16 + hi * 8];
    ra[kk] = a.v; ina[kk] = b.v;
  }
  v8f aor[4], asub[4], aoi[4];
  #pragma unroll
  for (int n = 0; n < 4; ++n) { aor[n] = zero8(); asub[n] = zero8(); aoi[n] = zero8(); }
  #pragma unroll
  for (int kk = 0; kk < 2; ++kk)
    #pragma unroll
    for (int n = 0; n < 4; ++n) {
      const int blk = (kk * 4 + n) * 512;
      v16bf bR = loadB(ws + ROP_OFF + blk, lane);
      v16bf bI = loadB(ws + IOP_OFF + blk, lane);
      aor[n]  = WMMA_BF16(ra[kk],  bR, aor[n]);
      asub[n] = WMMA_BF16(ina[kk], bI, asub[n]);
      aoi[n]  = WMMA_BF16(ra[kk],  bI, aoi[n]);
      aoi[n]  = WMMA_BF16(ina[kk], bR, aoi[n]);
    }
  #pragma unroll
  for (int n = 0; n < 4; ++n)
    #pragma unroll
    for (int i = 0; i < 8; ++i) {
      size_t ro = (size_t)(rowBase + i + 8 * hi) * Q + n * 16 + l15;
      __builtin_nontemporal_store(aor[n][i] - asub[n][i], &out[ro]);
      __builtin_nontemporal_store(aoi[n][i], &out[TOTAL + ro]);
    }
}

extern "C" void kernel_launch(void* const* d_in, const int* in_sizes, int n_in,
                              void* d_out, int out_size, void* d_ws, size_t ws_size,
                              hipStream_t stream) {
  const float* feat = (const float*)d_in[0];
  const float* Wr   = (const float*)d_in[1];
  const float* br   = (const float*)d_in[2];
  const float* Wi   = (const float*)d_in[3];
  const float* bi   = (const float*)d_in[4];
  const float* Wp1  = (const float*)d_in[5];
  const float* bp1  = (const float*)d_in[6];
  const float* Wp2  = (const float*)d_in[7];
  const float* bp2  = (const float*)d_in[8];
  const float* op   = (const float*)d_in[9];
  unsigned short* ws = (unsigned short*)d_ws;   // needs 221,184 bytes of scratch
  float* out = (float*)d_out;

  qi_prep<<<(WS_ELEMS + 255) / 256, 256, 0, stream>>>(Wr, Wi, Wp1, Wp2, op, ws);
  qi_main<<<NROWS / 128, 256, 0, stream>>>(feat, br, bi, bp1, bp2, ws, out);
}
